// CrossModalAttention_64630667870371
// MI455X (gfx1250) — matrix operand order repaired and checked by reading, not compiled, not run on hardware
//
#include <hip/hip_runtime.h>

#define BATCH 8
#define CDIM  256
#define QKDIM 32
#define NDIM  4096

typedef __bf16 bf16;
typedef __attribute__((ext_vector_type(16))) __bf16 v16bf;
typedef __attribute__((ext_vector_type(8)))  __bf16 v8bf;
typedef __attribute__((ext_vector_type(8)))  float  v8f;

#if defined(__has_builtin)
#if __has_builtin(__builtin_amdgcn_global_load_async_to_lds_b128) && \
    __has_builtin(__builtin_amdgcn_s_wait_asynccnt)
#define HAVE_ASYNC_LDS 1
#endif
#endif

#ifdef HAVE_ASYNC_LDS
// Builtin signature (from compiler diagnostic): param 1 is a global-AS
// (__device__) pointer to 'int __attribute__((vector_size(16)))'.
typedef int v4i_ __attribute__((vector_size(16)));
typedef __attribute__((address_space(1))) v4i_ g_v4i;
typedef __attribute__((address_space(3))) v4i_ l_v4i;
#endif

static __device__ __forceinline__ v16bf cat16(v8bf lo, v8bf hi) {
  v16bf a;
#pragma unroll
  for (int i = 0; i < 8; ++i) { a[i] = lo[i]; a[i + 8] = hi[i]; }
  return a;
}

static __device__ __forceinline__ v16bf cvt16(float4 a, float4 b, float4 c, float4 d) {
  v16bf r;
  r[0] = (bf16)a.x;  r[1] = (bf16)a.y;  r[2] = (bf16)a.z;  r[3] = (bf16)a.w;
  r[4] = (bf16)b.x;  r[5] = (bf16)b.y;  r[6] = (bf16)b.z;  r[7] = (bf16)b.w;
  r[8] = (bf16)c.x;  r[9] = (bf16)c.y;  r[10] = (bf16)c.z; r[11] = (bf16)c.w;
  r[12] = (bf16)d.x; r[13] = (bf16)d.y; r[14] = (bf16)d.z; r[15] = (bf16)d.w;
  return r;
}

// ---------------------------------------------------------------------------
// Kernel 1: QKV projection. Per block: one batch, one 64-wide n-tile.
//   q[d,n] = Wq[d,:]·x[:,n] + bq   -> stored transposed qT[b][n][32] (bf16)
//   k[d,n] likewise -> kT[b][n][32]
//   v[c,n] = Wv[c,:]·x[:,n] + bv   -> stored v[b][c][n] (bf16)
// ---------------------------------------------------------------------------
__global__ __launch_bounds__(256) void qkv_proj_wmma(
    const float* __restrict__ x,
    const float* __restrict__ Wq, const float* __restrict__ bq,
    const float* __restrict__ Wk, const float* __restrict__ bk,
    const float* __restrict__ Wv, const float* __restrict__ bv,
    bf16* __restrict__ qT, bf16* __restrict__ kT, bf16* __restrict__ vB)
{
  __shared__ __align__(16) bf16 xs[64 * CDIM];  // x tile transposed [n][c], 32 KB
  const int b   = blockIdx.y;
  const int n0  = blockIdx.x * 64;
  const int tid = threadIdx.x;

  // Stage x[c][n0..n0+63] -> LDS bf16 [n][c] (coalesced float4 along n).
  {
    const int cr = tid >> 4;          // 0..15
    const int n4 = (tid & 15) * 4;    // 0,4,...,60
#pragma unroll
    for (int it = 0; it < 16; ++it) {
      const int c = it * 16 + cr;
      const float4 xv = *(const float4*)(x + ((size_t)b * CDIM + c) * NDIM + n0 + n4);
      xs[(n4 + 0) * CDIM + c] = (bf16)xv.x;
      xs[(n4 + 1) * CDIM + c] = (bf16)xv.y;
      xs[(n4 + 2) * CDIM + c] = (bf16)xv.z;
      xs[(n4 + 3) * CDIM + c] = (bf16)xv.w;
    }
  }
  __syncthreads();

  const int wave = tid >> 5;
  const int lane = tid & 31;
  const int h    = lane >> 4;   // lane half (K-group select in WMMA layouts)
  const int r16  = lane & 15;

  // 80 output jobs: tiles 0-1 = q (d 0-15 / 16-31), 2-3 = k, 4-19 = v, x4 n-subtiles
  for (int job = wave; job < 80; job += 8) {
    const int tile = job >> 2;
    const int nsub = job & 3;
    v8f acc = {};

    if (tile < 4) {
      // q/k: D tile M=d, N=n. A = W rows (global), B = x^T columns (LDS).
      const bool  isQ = (tile < 2);
      const float* Wp = isQ ? Wq : Wk;
      const float* bp = isQ ? bq : bk;
      const int dbase = (tile & 1) * 16;
      const int drow  = dbase + r16;
      const int ncol  = nsub * 16 + r16;
#pragma unroll
      for (int kc = 0; kc < 8; ++kc) {
        const float* wr = Wp + (size_t)drow * CDIM + kc * 32;
        const v16bf A = cvt16(*(const float4*)(wr + h * 16),
                              *(const float4*)(wr + h * 16 + 4),
                              *(const float4*)(wr + h * 16 + 8),
                              *(const float4*)(wr + h * 16 + 12));
        const v8bf* bl = (const v8bf*)&xs[ncol * CDIM + kc * 32 + h * 16];
        const v16bf Bf = cat16(bl[0], bl[1]);
        acc = __builtin_amdgcn_wmma_f32_16x16x32_bf16(false, A, false, Bf,
                                                      (short)0, acc, false, false);
      }
      // rows d = dbase + r + 8h are contiguous over r -> packed bf16 store
      const float4 bb0 = *(const float4*)(bp + dbase + 8 * h);
      const float4 bb1 = *(const float4*)(bp + dbase + 8 * h + 4);
      const float bias[8] = {bb0.x, bb0.y, bb0.z, bb0.w, bb1.x, bb1.y, bb1.z, bb1.w};
      v8bf ov;
#pragma unroll
      for (int r = 0; r < 8; ++r) ov[r] = (bf16)(acc[r] + bias[r]);
      bf16* dst = (isQ ? qT : kT) +
                  ((size_t)b * NDIM + n0 + nsub * 16 + r16) * QKDIM + dbase + 8 * h;
      *(v8bf*)dst = ov;
    } else {
      // v: D tile M=n, N=c_out. A = x^T rows (LDS), B = Wv rows col-major (global).
      const int cbase = (tile - 4) * 16;
      const int cout  = cbase + r16;
      const int nrow  = nsub * 16 + r16;
#pragma unroll
      for (int kc = 0; kc < 8; ++kc) {
        const v8bf alo = *(const v8bf*)&xs[nrow * CDIM + kc * 32 + h * 16];
        const v8bf ahi = *(const v8bf*)&xs[nrow * CDIM + kc * 32 + h * 16 + 8];
        const v16bf A  = cat16(alo, ahi);
        const float* wr = Wv + (size_t)cout * CDIM + kc * 32 + h * 16;
        const v16bf Bf = cvt16(*(const float4*)(wr),     *(const float4*)(wr + 4),
                               *(const float4*)(wr + 8), *(const float4*)(wr + 12));
        acc = __builtin_amdgcn_wmma_f32_16x16x32_bf16(false, A, false, Bf,
                                                      (short)0, acc, false, false);
      }
      const float bvv = bv[cout];
      v8bf ov;
#pragma unroll
      for (int r = 0; r < 8; ++r) ov[r] = (bf16)(acc[r] + bvv);
      // rows n = n0 + nsub*16 + r + 8h contiguous over r -> packed store
      bf16* dst = vB + ((size_t)b * CDIM + cout) * NDIM + n0 + nsub * 16 + 8 * h;
      *(v8bf*)dst = ov;
    }
  }
}

// ---------------------------------------------------------------------------
// Kernel 2: flash attention. 4 waves/block; each wave owns 16 query rows and
// all 256 output channels. Keys/values stream in 32-wide steps with online
// softmax. k/v tiles are staged block-wide into LDS with double-buffered
// GLOBAL_LOAD_ASYNC_TO_LDS_B128 (ASYNCcnt), removing the 4x redundant
// per-wave global reads.
// ---------------------------------------------------------------------------
#ifdef HAVE_ASYNC_LDS
// Issue one step's tile DMA: k tile 32x32 bf16 (2 KB, 1 x b128/thread),
// v tile 256x32 bf16 (16 KB, 8 x b128/thread). 9 async ops per thread.
static __device__ __forceinline__ void issue_tile_loads(
    const bf16* kTb, const bf16* vBb, int m0, bf16* kdst, bf16* vdst, int tid)
{
  const char* ksrc = (const char*)(kTb + (size_t)m0 * QKDIM) + tid * 16;
  __builtin_amdgcn_global_load_async_to_lds_b128(
      (g_v4i*)(void*)ksrc, (l_v4i*)(void*)((char*)kdst + tid * 16), 0, 0);
#pragma unroll
  for (int j = 0; j < 8; ++j) {
    const int cid  = tid + j * 128;          // 0..1023 chunk id
    const int c    = cid >> 2;               // channel row
    const int part = cid & 3;                // 16B chunk within 64B row
    const char* vsrc = (const char*)(vBb + (size_t)c * NDIM + m0) + part * 16;
    char*       vd   = (char*)vdst + c * 64 + part * 16;
    __builtin_amdgcn_global_load_async_to_lds_b128(
        (g_v4i*)(void*)vsrc, (l_v4i*)(void*)vd, 0, 0);
  }
}
#endif

__global__ __launch_bounds__(128) void flash_attn_wmma(
    const float* __restrict__ x, const float* __restrict__ gamma_p,
    const bf16* __restrict__ qT, const bf16* __restrict__ kT,
    const bf16* __restrict__ vB, float* __restrict__ out)
{
  __shared__ __align__(16) bf16 Pls[4][16 * 32];   // per-wave P staging, 4 KB
#ifdef HAVE_ASYNC_LDS
  __shared__ __align__(16) bf16 kls[2][32 * QKDIM]; // double-buffered k tile, 4 KB
  __shared__ __align__(16) bf16 vls[2][CDIM * 32];  // double-buffered v tile, 32 KB
#endif
  const int b    = blockIdx.y;
  const int tid  = threadIdx.x;
  const int wave = tid >> 5;
  const int lane = tid & 31;
  const int h    = lane >> 4;
  const int r16  = lane & 15;
  const int n0   = blockIdx.x * 64 + wave * 16;
  const float gamma = gamma_p[0];

  // q A-fragment: rows n0+r16, full K=d=0..31
  const bf16* qrow = qT + ((size_t)b * NDIM + n0 + r16) * QKDIM;
  const v16bf qA = cat16(*(const v8bf*)(qrow + h * 16),
                         *(const v8bf*)(qrow + h * 16 + 8));

  float mrow[8], lrow[8];
  v8f acc[16];
#pragma unroll
  for (int r = 0; r < 8; ++r) { mrow[r] = -1e30f; lrow[r] = 0.0f; }
#pragma unroll
  for (int t = 0; t < 16; ++t) acc[t] = (v8f){};

  bf16* pw = &Pls[wave][0];
  const v8f zero = {};

  const bf16* kTb = kT + (size_t)b * NDIM * QKDIM;
  const bf16* vBb = vB + (size_t)b * CDIM * NDIM;

#ifdef HAVE_ASYNC_LDS
  issue_tile_loads(kTb, vBb, 0, kls[0], vls[0], tid);
#endif

  const int NSTEP = NDIM / 32;
  for (int s = 0; s < NSTEP; ++s) {
    const int m0  = s * 32;
#ifdef HAVE_ASYNC_LDS
    const int cur = s & 1;
    if (s + 1 < NSTEP) {
      issue_tile_loads(kTb, vBb, m0 + 32, kls[cur ^ 1], vls[cur ^ 1], tid);
      __builtin_amdgcn_s_wait_asynccnt(9);   // step-s tile landed; 9 next-step ops in flight
    } else {
      __builtin_amdgcn_s_wait_asynccnt(0);
    }
    __syncthreads();                          // tile visible to all 4 waves
    const bf16* k0 = &kls[cur][(size_t)r16 * QKDIM];
    const bf16* k1 = &kls[cur][(size_t)(16 + r16) * QKDIM];
#else
    const bf16* k0 = kTb + (size_t)(m0 + r16) * QKDIM;
    const bf16* k1 = k0 + 16 * QKDIM;
    if (m0 + 32 < NDIM) __builtin_prefetch((const void*)(k0 + 32 * QKDIM), 0, 1);
#endif
    // ---- S = q·kᵀ : two 16x16 tiles (keys m0..m0+31)
    const v16bf kB0 = cat16(*(const v8bf*)(k0 + h * 16), *(const v8bf*)(k0 + h * 16 + 8));
    const v16bf kB1 = cat16(*(const v8bf*)(k1 + h * 16), *(const v8bf*)(k1 + h * 16 + 8));
    v8f s0 = __builtin_amdgcn_wmma_f32_16x16x32_bf16(false, qA, false, kB0,
                                                     (short)0, zero, false, false);
    v8f s1 = __builtin_amdgcn_wmma_f32_16x16x32_bf16(false, qA, false, kB1,
                                                     (short)0, zero, false, false);

    // ---- online softmax (row r+8h lives in the lane's 16-lane half)
    float sc[8];
#pragma unroll
    for (int r = 0; r < 8; ++r) {
      float t = fmaxf(s0[r], s1[r]);
      t = fmaxf(t, __shfl_xor(t, 1, 32));
      t = fmaxf(t, __shfl_xor(t, 2, 32));
      t = fmaxf(t, __shfl_xor(t, 4, 32));
      t = fmaxf(t, __shfl_xor(t, 8, 32));
      const float mnew = fmaxf(mrow[r], t);
      sc[r]  = __expf(mrow[r] - mnew);
      mrow[r] = mnew;
      const float p0 = __expf(s0[r] - mnew);
      const float p1 = __expf(s1[r] - mnew);
      float ssum = p0 + p1;
      ssum += __shfl_xor(ssum, 1, 32);
      ssum += __shfl_xor(ssum, 2, 32);
      ssum += __shfl_xor(ssum, 4, 32);
      ssum += __shfl_xor(ssum, 8, 32);
      lrow[r] = lrow[r] * sc[r] + ssum;
      pw[(r + 8 * h) * 32 + r16]      = (bf16)p0;   // stage P for A-layout reload
      pw[(r + 8 * h) * 32 + 16 + r16] = (bf16)p1;
    }
    // rescale running accumulators
#pragma unroll
    for (int t = 0; t < 16; ++t)
#pragma unroll
      for (int r = 0; r < 8; ++r) acc[t][r] *= sc[r];

    // ---- P A-fragment (rows r16, K = local m 0..31); same-wave LDS is in-order
    const bf16* prow = pw + r16 * 32;
    const v16bf pA = cat16(*(const v8bf*)(prow + h * 16),
                           *(const v8bf*)(prow + h * 16 + 8));

    // ---- acc += P · V over 16 channel tiles
#pragma unroll
    for (int t = 0; t < 16; ++t) {
#ifdef HAVE_ASYNC_LDS
      const bf16* vp = &vls[cur][(size_t)(t * 16 + r16) * 32 + h * 16];
#else
      const bf16* vp = vBb + (size_t)(t * 16 + r16) * NDIM + m0 + h * 16;
#endif
      const v16bf vBf = cat16(*(const v8bf*)(vp), *(const v8bf*)(vp + 8));
      acc[t] = __builtin_amdgcn_wmma_f32_16x16x32_bf16(false, pA, false, vBf,
                                                       (short)0, acc[t], false, false);
    }
#ifdef HAVE_ASYNC_LDS
    __syncthreads();   // all waves done with buf[cur] before step s+2 overwrites it
#endif
  }

  // ---- epilogue: out[b][c][n] = gamma * acc/l + x ; rows contiguous per lane
  float inv[8];
#pragma unroll
  for (int r = 0; r < 8; ++r) inv[r] = gamma / lrow[r];
#pragma unroll
  for (int t = 0; t < 16; ++t) {
    const int c = t * 16 + r16;
    const size_t base = ((size_t)b * CDIM + c) * NDIM + n0 + 8 * h;
    const float4 x0 = *(const float4*)(x + base);
    const float4 x1 = *(const float4*)(x + base + 4);
    float4 o0, o1;
    o0.x = acc[t][0] * inv[0] + x0.x;
    o0.y = acc[t][1] * inv[1] + x0.y;
    o0.z = acc[t][2] * inv[2] + x0.z;
    o0.w = acc[t][3] * inv[3] + x0.w;
    o1.x = acc[t][4] * inv[4] + x1.x;
    o1.y = acc[t][5] * inv[5] + x1.y;
    o1.z = acc[t][6] * inv[6] + x1.z;
    o1.w = acc[t][7] * inv[7] + x1.w;
    *(float4*)(out + base)     = o0;
    *(float4*)(out + base + 4) = o1;
  }
}

extern "C" void kernel_launch(void* const* d_in, const int* in_sizes, int n_in,
                              void* d_out, int out_size, void* d_ws, size_t ws_size,
                              hipStream_t stream) {
  const float* x  = (const float*)d_in[0];
  const float* Wq = (const float*)d_in[1];
  const float* bq = (const float*)d_in[2];
  const float* Wk = (const float*)d_in[3];
  const float* bk = (const float*)d_in[4];
  const float* Wv = (const float*)d_in[5];
  const float* bv = (const float*)d_in[6];
  const float* gm = (const float*)d_in[7];
  float* out = (float*)d_out;

  // workspace: qT (2 MB) | kT (2 MB) | v bf16 (16 MB)
  bf16* qT = (bf16*)d_ws;
  bf16* kT = qT + (size_t)BATCH * NDIM * QKDIM;
  bf16* vB = kT + (size_t)BATCH * NDIM * QKDIM;

  qkv_proj_wmma<<<dim3(NDIM / 64, BATCH), 256, 0, stream>>>(
      x, Wq, bq, Wk, bk, Wv, bv, qT, kT, vB);
  flash_attn_wmma<<<dim3(NDIM / 64, BATCH), 128, 0, stream>>>(
      x, gm, qT, kT, vB, out);
}
